// DifferentialWindowAttention_59133109732044
// MI455X (gfx1250) — compile-verified, matrix operand order checked by reading, not run on hardware
//
#include <hip/hip_runtime.h>
#include <hip/hip_bf16.h>
#include <stdint.h>

// ---------------------------------------------------------------------------
// DifferentialWindowAttention for MI455X (gfx1250), all GEMMs via
// v_wmma_f32_16x16x32_bf16 (wave32 WMMA, f32 accumulate), native bf16 cvt,
// software-pipelined (double-buffered) GEMM k-loops for load/WMMA overlap.
// ---------------------------------------------------------------------------

typedef __bf16 bf16_t;
typedef __attribute__((ext_vector_type(16))) __bf16 v16bf;
typedef __attribute__((ext_vector_type(8)))  float  v8f;

#define DEVINL __device__ __forceinline__

constexpr int kB    = 2048;
constexpr int kN    = 49;
constexpr int kDIM  = 512;
constexpr int kH    = 16;
constexpr int kHD   = 32;
constexpr int kLHD  = 8;
constexpr int kLDIM = 128;
constexpr int kQKVL = 1920;              // 1536 qkv + 3*128 lite, concatenated
constexpr int kRows = kB * kN;           // 100352 (= 6272 * 16)

union FragU { uint32_t u[8]; v16bf v; };

DEVINL bf16_t f2bf(float f) { return (bf16_t)f; }          // native v_cvt
DEVINL float  bf2f(bf16_t h) { return (float)h; }
DEVINL uint32_t packbf(float a, float b) {                 // v_cvt_pk_bf16_f32
  union { bf16_t h[2]; uint32_t u; } r;
  r.h[0] = (bf16_t)a;
  r.h[1] = (bf16_t)b;
  return r.u;
}

DEVINL v16bf frag_zero() {
  FragU r;
#pragma unroll
  for (int i = 0; i < 8; i++) r.u[i] = 0u;
  return r.v;
}

// 16 contiguous bf16 (32 bytes, 16B aligned) -> fragment (pure b128 loads)
DEVINL v16bf frag_load_bf16(const bf16_t* p) {
  FragU r;
  const uint4* q = (const uint4*)p;
  uint4 a = q[0];
  uint4 b = q[1];
  r.u[0] = a.x; r.u[1] = a.y; r.u[2] = a.z; r.u[3] = a.w;
  r.u[4] = b.x; r.u[5] = b.y; r.u[6] = b.z; r.u[7] = b.w;
  return r.v;
}

DEVINL v8f wmma_bf16(v16bf a, v16bf b, v8f c) {
  return __builtin_amdgcn_wmma_f32_16x16x32_bf16(
      /*neg_a=*/false, a, /*neg_b=*/false, b,
      /*c_mod=*/(short)0, c, /*reuse_a=*/false, /*reuse_b=*/false);
}

// Row softmax over 64 columns held as 4 col-tiles x 8 row-slots per lane.
DEVINL void softmax64(float (&l)[4][8]) {
#pragma unroll
  for (int i = 0; i < 8; i++) {
    float mx = l[0][i];
#pragma unroll
    for (int c = 1; c < 4; c++) mx = fmaxf(mx, l[c][i]);
#pragma unroll
    for (int off = 1; off < 16; off <<= 1) mx = fmaxf(mx, __shfl_xor(mx, off, 16));
    float s = 0.0f;
#pragma unroll
    for (int c = 0; c < 4; c++) { float e = __expf(l[c][i] - mx); l[c][i] = e; s += e; }
#pragma unroll
    for (int off = 1; off < 16; off <<= 1) s += __shfl_xor(s, off, 16);
    float inv = 1.0f / s;
#pragma unroll
    for (int c = 0; c < 4; c++) l[c][i] *= inv;
  }
}

// ---------------------------------------------------------------------------
// Kernel 0a: weights -> bf16; concatenated qkv+lite weight and bias vector.
// ---------------------------------------------------------------------------
__global__ void __launch_bounds__(256)
k_prep_weights(const float* __restrict__ qkv_w, const float* __restrict__ qlw,
               const float* __restrict__ klw,   const float* __restrict__ vlw,
               const float* __restrict__ proj_w, const float* __restrict__ dpw,
               const float* __restrict__ qkv_b,
               bf16_t* __restrict__ wcat,  bf16_t* __restrict__ wproj,
               bf16_t* __restrict__ wdiff, float* __restrict__ biascat) {
  int idx = blockIdx.x * blockDim.x + threadIdx.x;
  if (idx < kQKVL * kDIM) {
    int r = idx / kDIM, c = idx % kDIM;
    float v;
    if      (r < 1536) v = qkv_w[idx];
    else if (r < 1664) v = qlw[(r - 1536) * kDIM + c];
    else if (r < 1792) v = klw[(r - 1664) * kDIM + c];
    else               v = vlw[(r - 1792) * kDIM + c];
    wcat[idx] = f2bf(v);
  }
  if (idx < kDIM * kDIM)  wproj[idx] = f2bf(proj_w[idx]);
  if (idx < kDIM * kLDIM) wdiff[idx] = f2bf(dpw[idx]);
  if (idx < kQKVL)        biascat[idx] = (idx < 1536) ? qkv_b[idx] : 0.0f;
}

// ---------------------------------------------------------------------------
// Kernel 0b: x (f32) -> xb (bf16), 8 elements / thread, b128 in / b128 out.
// ---------------------------------------------------------------------------
__global__ void __launch_bounds__(256)
k_cvt_x(const float* __restrict__ x, bf16_t* __restrict__ xb) {
  const size_t i = ((size_t)blockIdx.x * 256 + threadIdx.x) * 8;
  const float4* p = (const float4*)(x + i);
  float4 f0 = p[0];
  float4 f1 = p[1];
  uint4 o;
  o.x = packbf(f0.x, f0.y);
  o.y = packbf(f0.z, f0.w);
  o.z = packbf(f1.x, f1.y);
  o.w = packbf(f1.z, f1.w);
  *(uint4*)(xb + i) = o;
}

// ---------------------------------------------------------------------------
// Kernel 1: qkv+lite projection GEMM, 16x64 per wave, double-buffered K loop.
// C[100352 x 1920](bf16) = xb @ wcat^T + biascat
// ---------------------------------------------------------------------------
__global__ void __launch_bounds__(256)
k_gemm_qkvl(const bf16_t* __restrict__ xb, const bf16_t* __restrict__ wcat,
            const float* __restrict__ biascat, bf16_t* __restrict__ qkvl) {
  const int wave = threadIdx.x >> 5, lane = threadIdx.x & 31;
  const int hl = lane >> 4, l16 = lane & 15;
  const int NTQ = kQKVL / 64;                 // 30 quads of N tiles
  const int tile = blockIdx.x * 8 + wave;
  const int tm = tile / NTQ, tq = tile % NTQ;
  const int tn0 = tq * 4;

  const bf16_t* arow = xb + (size_t)(tm * 16 + l16) * kDIM + hl * 16;
  const bf16_t* brow[4];
#pragma unroll
  for (int j = 0; j < 4; j++)
    brow[j] = wcat + (size_t)((tn0 + j) * 16 + l16) * kDIM + hl * 16;

  v8f acc[4];
#pragma unroll
  for (int j = 0; j < 4; j++) acc[j] = (v8f){0.f,0.f,0.f,0.f,0.f,0.f,0.f,0.f};

  // prologue: fragments for k0 = 0
  v16bf a_cur = frag_load_bf16(arow);
  v16bf b_cur[4];
#pragma unroll
  for (int j = 0; j < 4; j++) b_cur[j] = frag_load_bf16(brow[j]);

#pragma unroll 4
  for (int k0 = 0; k0 < kDIM; k0 += 32) {
    const int kp = (k0 + 32 < kDIM) ? (k0 + 32) : 0;   // clamp: dummy re-load
    v16bf a_nxt = frag_load_bf16(arow + kp);
    v16bf b_nxt[4];
#pragma unroll
    for (int j = 0; j < 4; j++) b_nxt[j] = frag_load_bf16(brow[j] + kp);
#pragma unroll
    for (int j = 0; j < 4; j++) acc[j] = wmma_bf16(a_cur, b_cur[j], acc[j]);
    a_cur = a_nxt;
#pragma unroll
    for (int j = 0; j < 4; j++) b_cur[j] = b_nxt[j];
  }

#pragma unroll
  for (int j = 0; j < 4; j++) {
    const int col = (tn0 + j) * 16 + l16;
    const float bias = biascat[col];
#pragma unroll
    for (int i = 0; i < 8; i++) {
      const size_t row = (size_t)(tm * 16 + i + 8 * hl);
      qkvl[row * kQKVL + col] = f2bf(acc[j][i] + bias);
    }
  }
}

// ---------------------------------------------------------------------------
// Kernel 2: per-(window, head) differential attention.
// ---------------------------------------------------------------------------
__global__ void __launch_bounds__(128)
k_attention(const bf16_t* __restrict__ qkvl,
            const float* __restrict__ mask, const int* __restrict__ relidx,
            const float* __restrict__ rpb,
            const float* __restrict__ lam1p, const float* __restrict__ lam2p,
            bf16_t* __restrict__ xorig, bf16_t* __restrict__ dv) {
  __shared__ __align__(16) float  bias_s[64 * 64];   // rpb+mask logit bias, -1e30 pads
  __shared__ __align__(16) bf16_t vT_s[32 * 64];     // v transposed  [d][t]
  __shared__ __align__(16) bf16_t vlT_s[16 * 64];    // vl transposed [d][t], d>=8 zero
  __shared__ __align__(16) bf16_t p0_s[4][16 * 64];  // per-wave prob tile (orig)
  __shared__ __align__(16) bf16_t p1_s[4][16 * 64];  // per-wave prob tile (diff)

  const int blk = blockIdx.x;
  const int b = blk >> 4;
  const int h = blk & 15;
  const int w = b & 63;                    // window index within mask
  const int tid = threadIdx.x;

  // ---- cooperative LDS fill ----
  for (int e = tid; e < 64 * 64; e += 128) {
    int m = e >> 6, n = e & 63;
    float v = -1e30f;
    if (m < kN && n < kN)
      v = rpb[relidx[m * kN + n] * kH + h] + mask[((size_t)w * kN + m) * kN + n];
    bias_s[e] = v;
  }
  for (int e = tid; e < 32 * 64; e += 128) {
    int d = e >> 6, t = e & 63;
    bf16_t v = (bf16_t)0.0f;
    if (t < kN) v = qkvl[(size_t)(b * kN + t) * kQKVL + 1024 + h * kHD + d];
    vT_s[e] = v;
  }
  for (int e = tid; e < 16 * 64; e += 128) {
    int d = e >> 6, t = e & 63;
    bf16_t v = (bf16_t)0.0f;
    if (d < kLHD && t < kN) v = qkvl[(size_t)(b * kN + t) * kQKVL + 1792 + h * kLHD + d];
    vlT_s[e] = v;
  }
  __syncthreads();

  const int wave = tid >> 5, lane = tid & 31;
  const int hl = lane >> 4, l16 = lane & 15;
  const int r = wave;
  const int arow = 16 * r + l16;
  const bool aok = arow < kN;
  const size_t rowbase = (size_t)(b * kN + (aok ? arow : 0)) * kQKVL;
  const v8f vzero = {0.f, 0.f, 0.f, 0.f, 0.f, 0.f, 0.f, 0.f};

  // ---- dense logits q @ k^T (K = hd = 32 -> one WMMA per column tile) ----
  float lg[4][8];
  {
    v16bf aq = aok ? frag_load_bf16(qkvl + rowbase + h * kHD + hl * 16) : frag_zero();
#pragma unroll
    for (int c = 0; c < 4; c++) {
      const int ncol = 16 * c + l16;
      v16bf bk = (ncol < kN)
          ? frag_load_bf16(qkvl + (size_t)(b * kN + ncol) * kQKVL + 512 + h * kHD + hl * 16)
          : frag_zero();
      v8f acc = wmma_bf16(aq, bk, vzero);
#pragma unroll
      for (int i = 0; i < 8; i++)
        lg[c][i] = acc[i] * 0.17677669529663687f            // 32^-0.5
                 + bias_s[(16 * r + i + 8 * hl) * 64 + 16 * c + l16];
    }
  }
  softmax64(lg);                                            // lg := attn_orig probs

#pragma unroll
  for (int c = 0; c < 4; c++)
#pragma unroll
    for (int i = 0; i < 8; i++)
      p0_s[wave][(i + 8 * hl) * 64 + 16 * c + l16] = f2bf(lg[c][i]);

  // ---- lite logits ql @ kl^T (K = 8, zero-padded bf16 fragments) ----
  float lgl[4][8];
  {
    v16bf aql = frag_zero();
    if (aok && hl == 0) {
      FragU rr;
      uint4 t = *(const uint4*)(qkvl + rowbase + 1536 + h * kLHD);
      rr.u[0] = t.x; rr.u[1] = t.y; rr.u[2] = t.z; rr.u[3] = t.w;
      rr.u[4] = 0;   rr.u[5] = 0;   rr.u[6] = 0;   rr.u[7] = 0;
      aql = rr.v;
    }
#pragma unroll
    for (int c = 0; c < 4; c++) {
      const int ncol = 16 * c + l16;
      v16bf bkl = frag_zero();
      if (ncol < kN && hl == 0) {
        FragU rr;
        uint4 t = *(const uint4*)(qkvl + (size_t)(b * kN + ncol) * kQKVL + 1664 + h * kLHD);
        rr.u[0] = t.x; rr.u[1] = t.y; rr.u[2] = t.z; rr.u[3] = t.w;
        rr.u[4] = 0;   rr.u[5] = 0;   rr.u[6] = 0;   rr.u[7] = 0;
        bkl = rr.v;
      }
      v8f acc = wmma_bf16(aql, bkl, vzero);
#pragma unroll
      for (int i = 0; i < 8; i++)
        lgl[c][i] = acc[i] * 0.3535533905932738f            // 8^-0.5
                  + bias_s[(16 * r + i + 8 * hl) * 64 + 16 * c + l16];
    }
  }
  softmax64(lgl);                                           // lgl := attn_lite probs

  const float l1 = 1.0f / (1.0f + __expf(-lam1p[0]));
  const float l2 = 1.0f / (1.0f + __expf(-lam2p[0]));
#pragma unroll
  for (int c = 0; c < 4; c++)
#pragma unroll
    for (int i = 0; i < 8; i++)
      p1_s[wave][(i + 8 * hl) * 64 + 16 * c + l16] =
          f2bf(l1 * lgl[c][i] - l2 * lg[c][i]);

  // ---- p0 @ v  and  p1 @ vl  (K = 64 tokens -> 2 WMMA steps) ----
  v8f xo0 = vzero, xo1 = vzero, dva = vzero;
#pragma unroll
  for (int ks = 0; ks < 2; ks++) {
    const int ko = ks * 32 + hl * 16;
    v16bf ap0 = frag_load_bf16(&p0_s[wave][l16 * 64 + ko]);
    xo0 = wmma_bf16(ap0, frag_load_bf16(&vT_s[(l16)      * 64 + ko]), xo0);
    xo1 = wmma_bf16(ap0, frag_load_bf16(&vT_s[(16 + l16) * 64 + ko]), xo1);
    v16bf ap1 = frag_load_bf16(&p1_s[wave][l16 * 64 + ko]);
    dva = wmma_bf16(ap1, frag_load_bf16(&vlT_s[l16 * 64 + ko]), dva);
  }
#pragma unroll
  for (int i = 0; i < 8; i++) {
    const int t = 16 * r + i + 8 * hl;
    if (t < kN) {
      const size_t ob = (size_t)(b * kN + t) * kDIM + h * kHD;
      xorig[ob + l16]      = f2bf(xo0[i]);
      xorig[ob + 16 + l16] = f2bf(xo1[i]);
      if (l16 < kLHD)
        dv[(size_t)(b * kN + t) * kLDIM + h * kLHD + l16] = f2bf(dva[i]);
    }
  }
}

// ---------------------------------------------------------------------------
// Kernel 3: y = x_orig + gamma * (dv @ diff_proj^T + diff_b), bf16 in place.
// 16x64 per wave, double-buffered K loop (4 steps).
// ---------------------------------------------------------------------------
__global__ void __launch_bounds__(256)
k_gemm_diff(const bf16_t* __restrict__ dv, const bf16_t* __restrict__ wdiff,
            const float* __restrict__ dpb, const float* __restrict__ gammap,
            bf16_t* __restrict__ y /* holds x_orig, overwritten with y */) {
  const int wave = threadIdx.x >> 5, lane = threadIdx.x & 31;
  const int hl = lane >> 4, l16 = lane & 15;
  const int NTQ = kDIM / 64;                  // 8 quads
  const int tile = blockIdx.x * 8 + wave;
  const int tm = tile / NTQ, tq = tile % NTQ;
  const int tn0 = tq * 4;

  const bf16_t* arow = dv + (size_t)(tm * 16 + l16) * kLDIM + hl * 16;
  const bf16_t* brow[4];
#pragma unroll
  for (int j = 0; j < 4; j++)
    brow[j] = wdiff + (size_t)((tn0 + j) * 16 + l16) * kLDIM + hl * 16;

  v8f acc[4];
#pragma unroll
  for (int j = 0; j < 4; j++) acc[j] = (v8f){0.f,0.f,0.f,0.f,0.f,0.f,0.f,0.f};

  v16bf a_cur = frag_load_bf16(arow);
  v16bf b_cur[4];
#pragma unroll
  for (int j = 0; j < 4; j++) b_cur[j] = frag_load_bf16(brow[j]);

#pragma unroll
  for (int k0 = 0; k0 < kLDIM; k0 += 32) {
    const int kp = (k0 + 32 < kLDIM) ? (k0 + 32) : 0;
    v16bf a_nxt = frag_load_bf16(arow + kp);
    v16bf b_nxt[4];
#pragma unroll
    for (int j = 0; j < 4; j++) b_nxt[j] = frag_load_bf16(brow[j] + kp);
#pragma unroll
    for (int j = 0; j < 4; j++) acc[j] = wmma_bf16(a_cur, b_cur[j], acc[j]);
    a_cur = a_nxt;
#pragma unroll
    for (int j = 0; j < 4; j++) b_cur[j] = b_nxt[j];
  }

  const float g = gammap[0];
#pragma unroll
  for (int j = 0; j < 4; j++) {
    const int col = (tn0 + j) * 16 + l16;
    const float bias = dpb[col];
#pragma unroll
    for (int i = 0; i < 8; i++) {
      const size_t idx = (size_t)(tm * 16 + i + 8 * hl) * kDIM + col;
      y[idx] = f2bf(bf2f(y[idx]) + g * (acc[j][i] + bias));
    }
  }
}

// ---------------------------------------------------------------------------
// Kernel 4: out = y @ proj^T + proj_b (f32 out), 16x64 per wave, pipelined.
// ---------------------------------------------------------------------------
__global__ void __launch_bounds__(256)
k_gemm_out(const bf16_t* __restrict__ y, const bf16_t* __restrict__ wproj,
           const float* __restrict__ pb, float* __restrict__ out) {
  const int wave = threadIdx.x >> 5, lane = threadIdx.x & 31;
  const int hl = lane >> 4, l16 = lane & 15;
  const int NTQ = kDIM / 64;                  // 8 quads
  const int tile = blockIdx.x * 8 + wave;
  const int tm = tile / NTQ, tq = tile % NTQ;
  const int tn0 = tq * 4;

  const bf16_t* arow = y + (size_t)(tm * 16 + l16) * kDIM + hl * 16;
  const bf16_t* brow[4];
#pragma unroll
  for (int j = 0; j < 4; j++)
    brow[j] = wproj + (size_t)((tn0 + j) * 16 + l16) * kDIM + hl * 16;

  v8f acc[4];
#pragma unroll
  for (int j = 0; j < 4; j++) acc[j] = (v8f){0.f,0.f,0.f,0.f,0.f,0.f,0.f,0.f};

  v16bf a_cur = frag_load_bf16(arow);
  v16bf b_cur[4];
#pragma unroll
  for (int j = 0; j < 4; j++) b_cur[j] = frag_load_bf16(brow[j]);

#pragma unroll 4
  for (int k0 = 0; k0 < kDIM; k0 += 32) {
    const int kp = (k0 + 32 < kDIM) ? (k0 + 32) : 0;
    v16bf a_nxt = frag_load_bf16(arow + kp);
    v16bf b_nxt[4];
#pragma unroll
    for (int j = 0; j < 4; j++) b_nxt[j] = frag_load_bf16(brow[j] + kp);
#pragma unroll
    for (int j = 0; j < 4; j++) acc[j] = wmma_bf16(a_cur, b_cur[j], acc[j]);
    a_cur = a_nxt;
#pragma unroll
    for (int j = 0; j < 4; j++) b_cur[j] = b_nxt[j];
  }

#pragma unroll
  for (int j = 0; j < 4; j++) {
    const int col = (tn0 + j) * 16 + l16;
    const float bias = pb[col];
#pragma unroll
    for (int i = 0; i < 8; i++) {
      const size_t row = (size_t)(tm * 16 + i + 8 * hl);
      out[row * kDIM + col] = acc[j][i] + bias;
    }
  }
}

// ---------------------------------------------------------------------------
extern "C" void kernel_launch(void* const* d_in, const int* in_sizes, int n_in,
                              void* d_out, int out_size, void* d_ws, size_t ws_size,
                              hipStream_t stream) {
  (void)in_sizes; (void)n_in; (void)out_size; (void)ws_size;

  const float* x      = (const float*)d_in[0];
  const float* mask   = (const float*)d_in[1];
  const int*   relidx = (const int*)  d_in[2];
  const float* rpb    = (const float*)d_in[3];
  const float* qkv_w  = (const float*)d_in[4];
  const float* qkv_b  = (const float*)d_in[5];
  const float* proj_w = (const float*)d_in[6];
  const float* proj_b = (const float*)d_in[7];
  const float* qlw    = (const float*)d_in[8];
  const float* klw    = (const float*)d_in[9];
  const float* vlw    = (const float*)d_in[10];
  const float* lam1   = (const float*)d_in[11];
  const float* lam2   = (const float*)d_in[12];
  const float* dpw    = (const float*)d_in[13];
  const float* dpb    = (const float*)d_in[14];
  const float* gamma  = (const float*)d_in[15];

  auto AL = [](size_t v) { return (v + 255) & ~(size_t)255; };
  char* ws = (char*)d_ws;
  const size_t o_wcat  = 0;
  const size_t o_bias  = o_wcat  + AL((size_t)kQKVL * kDIM * 2);
  const size_t o_wproj = o_bias  + AL((size_t)kQKVL * 4);
  const size_t o_wdiff = o_wproj + AL((size_t)kDIM * kDIM * 2);
  const size_t o_xb    = o_wdiff + AL((size_t)kDIM * kLDIM * 2);
  const size_t o_qkvl  = o_xb    + AL((size_t)kRows * kDIM * 2);
  const size_t o_xorig = o_qkvl  + AL((size_t)kRows * kQKVL * 2);
  const size_t o_dv    = o_xorig + AL((size_t)kRows * kDIM * 2);

  bf16_t* wcat  = (bf16_t*)(ws + o_wcat);
  float*  bcat  = (float*) (ws + o_bias);
  bf16_t* wproj = (bf16_t*)(ws + o_wproj);
  bf16_t* wdiff = (bf16_t*)(ws + o_wdiff);
  bf16_t* xb    = (bf16_t*)(ws + o_xb);
  bf16_t* qkvl  = (bf16_t*)(ws + o_qkvl);
  bf16_t* xorig = (bf16_t*)(ws + o_xorig);
  bf16_t* dvb   = (bf16_t*)(ws + o_dv);

  // 0a) weight conversion (covers 1920*512 = 983040 elements)
  k_prep_weights<<<3840, 256, 0, stream>>>(qkv_w, qlw, klw, vlw, proj_w, dpw, qkv_b,
                                           wcat, wproj, wdiff, bcat);
  // 0b) x -> bf16: 51,380,224 elems / 8 per thread / 256 = 25088 blocks
  k_cvt_x<<<25088, 256, 0, stream>>>(x, xb);

  // 1) qkv + lite projections: 6272 * 30 tile-quads / 8 waves = 23520 blocks
  k_gemm_qkvl<<<23520, 256, 0, stream>>>(xb, wcat, bcat, qkvl);

  // 2) attention: one block per (window, head)
  k_attention<<<kB * kH, 128, 0, stream>>>(qkvl, mask, relidx, rpb, lam1, lam2,
                                           xorig, dvb);

  // 3) y = x_orig + gamma*(dv @ diff^T + db): 6272 * 8 / 8 = 6272 blocks
  k_gemm_diff<<<6272, 256, 0, stream>>>(dvb, wdiff, dpb, gamma, xorig);

  // 4) out = y @ proj^T + proj_b
  k_gemm_out<<<6272, 256, 0, stream>>>(xorig, wproj, proj_b, (float*)d_out);
}
